// InteractionBlock_6493990551741
// MI455X (gfx1250) — matrix operand order repaired and checked by reading, not compiled, not run on hardware
//
#include <hip/hip_runtime.h>

typedef __attribute__((ext_vector_type(16))) _Float16 v16h;
typedef __attribute__((ext_vector_type(8)))  float    v8f;

#define DIM 288
#define NRBF 8
#define HID 64
#define WCOLS 224   // N_PATHS * L0
#define EWAVES 2    // waves per edge block
#define EBLOCK (EWAVES * 32)

union V16H { v16h v; _Float16 e[16]; };

__device__ __forceinline__ float sigmoidf_(float x) { return 1.0f / (1.0f + __expf(-x)); }
__device__ __forceinline__ float siluf_(float x)    { return x * sigmoidf_(x); }

// ---------------------------------------------------------------------------
// Pack Wr1 (8x64, K zero-padded to 32) and Wr2 (64x224) into f16 WMMA
// B-fragment layout: fragment = [32 lanes][16 halves]; lane (hf=lane>>4,
// m=lane&15) element i holds B[K = kcBase + hf*16 + i][N = tile*16 + m].
// ---------------------------------------------------------------------------
__global__ void pack_kernel(const float* __restrict__ Wr1,
                            const float* __restrict__ Wr2,
                            _Float16* __restrict__ p1,
                            _Float16* __restrict__ p2) {
    int b = blockIdx.x;          // 0..31 : 4 Wr1 tiles + 28 Wr2 (kc,t) tiles
    int lane = threadIdx.x;      // 0..31
    int m = lane & 15, hf = lane >> 4;
    if (b < 4) {
        int t = b;
        _Float16* dst = p1 + ((size_t)(t * 32 + lane)) * 16;
        #pragma unroll
        for (int i = 0; i < 16; ++i) {
            int K = hf * 16 + i;
            int Ncol = t * 16 + m;
            dst[i] = (K < NRBF) ? (_Float16)Wr1[K * HID + Ncol] : (_Float16)0.0f;
        }
    } else {
        int q = b - 4;
        int kc = q / 14, t = q % 14;
        _Float16* dst = p2 + ((size_t)((kc * 14 + t) * 32 + lane)) * 16;
        #pragma unroll
        for (int i = 0; i < 16; ++i) {
            int K = kc * 32 + hf * 16 + i;       // 0..63
            int Ncol = t * 16 + m;               // 0..223
            dst[i] = (_Float16)Wr2[K * WCOLS + Ncol];
        }
    }
}

__global__ void zero_kernel(float* __restrict__ p, long long n) {
    long long i = (long long)blockIdx.x * blockDim.x + threadIdx.x;
    long long stride = (long long)gridDim.x * blockDim.x;
    for (; i < n; i += stride) p[i] = 0.0f;
}

// ---------------------------------------------------------------------------
// Shared helper: one node row (288) -> irreps_linear outputs, 9 per lane.
// row / w* live in LDS.
// ---------------------------------------------------------------------------
__device__ __forceinline__ void linear_row(const float* row,
                                           const float* w0, const float* w1,
                                           const float* w2, int lane,
                                           float out[9]) {
    #pragma unroll
    for (int k = 0; k < 9; ++k) {
        int j = lane + 32 * k;
        float acc = 0.0f;
        if (j < 32) {
            #pragma unroll
            for (int i = 0; i < 32; ++i) acc += row[i] * w0[i * 32 + j];
        } else if (j < 128) {
            int jm = j - 32; int c = jm % 3; int l = jm / 3;
            #pragma unroll
            for (int i = 0; i < 32; ++i) acc += row[32 + i * 3 + c] * w1[i * 32 + l];
        } else {
            int jm = j - 128; int c = jm % 5; int l = jm / 5;
            #pragma unroll
            for (int i = 0; i < 32; ++i) acc += row[128 + i * 5 + c] * w2[i * 32 + l];
        }
        out[k] = acc;
    }
}

// si1 = irreps_linear(nodes)
__global__ __launch_bounds__(256) void linear_si_kernel(
    const float* __restrict__ X, const float* __restrict__ W0,
    const float* __restrict__ W1, const float* __restrict__ W2,
    float* __restrict__ Y, int N) {
    __shared__ float sW0[1024], sW1[1024], sW2[1024];
    __shared__ float sRow[8][DIM];
    int tid = threadIdx.x;
    for (int i = tid; i < 1024; i += 256) { sW0[i] = W0[i]; sW1[i] = W1[i]; sW2[i] = W2[i]; }
    __syncthreads();
    int wv = tid >> 5, lane = tid & 31;
    int n = blockIdx.x * 8 + wv;
    if (n >= N) return;
    const float* x = X + (size_t)n * DIM;
    #pragma unroll
    for (int k = 0; k < 9; ++k) sRow[wv][lane + 32 * k] = x[lane + 32 * k];
    float outv[9];
    linear_row(sRow[wv], sW0, sW1, sW2, lane, outv);
    float* yr = Y + (size_t)n * DIM;
    #pragma unroll
    for (int k = 0; k < 9; ++k) yr[lane + 32 * k] = outv[k];
}

// out = gate(nodes + irreps_linear(conv))
__global__ __launch_bounds__(256) void linear_gate_kernel(
    const float* __restrict__ conv, const float* __restrict__ nodes,
    const float* __restrict__ W0, const float* __restrict__ W1,
    const float* __restrict__ W2, const float* __restrict__ Wg,
    float* __restrict__ out, int N) {
    __shared__ float sW0[1024], sW1[1024], sW2[1024], sWg[2048];
    __shared__ float sRow[8][DIM];
    int tid = threadIdx.x;
    for (int i = tid; i < 1024; i += 256) { sW0[i] = W0[i]; sW1[i] = W1[i]; sW2[i] = W2[i]; }
    for (int i = tid; i < 2048; i += 256) sWg[i] = Wg[i];
    __syncthreads();
    int wv = tid >> 5, lane = tid & 31;
    int n = blockIdx.x * 8 + wv;
    if (n >= N) return;
    const float* cr = conv + (size_t)n * DIM;
    #pragma unroll
    for (int k = 0; k < 9; ++k) sRow[wv][lane + 32 * k] = cr[lane + 32 * k];
    float outv[9];
    linear_row(sRow[wv], sW0, sW1, sW2, lane, outv);
    const float* nr = nodes + (size_t)n * DIM;
    #pragma unroll
    for (int k = 0; k < 9; ++k) outv[k] += nr[lane + 32 * k];   // mixed
    // stage mixed row (wave-lockstep: all reads of conv row are already done)
    #pragma unroll
    for (int k = 0; k < 9; ++k) sRow[wv][lane + 32 * k] = outv[k];
    float* orow = out + (size_t)n * DIM;
    #pragma unroll
    for (int k = 0; k < 9; ++k) {
        int j = lane + 32 * k;
        float v;
        if (j < 32) {
            v = siluf_(outv[k]);
        } else {
            int gi = (j < 128) ? (j - 32) / 3 : 32 + (j - 128) / 5;
            float acc = 0.0f;
            #pragma unroll
            for (int i = 0; i < 32; ++i) acc += sRow[wv][i] * sWg[i * 64 + gi];
            v = outv[k] * sigmoidf_(acc);
        }
        orow[j] = v;
    }
}

// ---------------------------------------------------------------------------
// Edge kernel: geometry + RBF -> WMMA MLP (8->64->224) -> messages -> atomic
// scatter into conv. One wave owns 16 edges and its own LDS slice (no
// cross-wave sharing -> no barriers; per-wave DS ordering covers reuse).
// ---------------------------------------------------------------------------
__global__ __launch_bounds__(EBLOCK) void edge_kernel(
    const float* __restrict__ si1, const float* __restrict__ pos,
    const int* __restrict__ eidx, const float* __restrict__ br1,
    const _Float16* __restrict__ p1, const _Float16* __restrict__ p2,
    float* __restrict__ conv, int E) {
    __shared__ float    sY[EWAVES][16][8];        // Y1[0..2], Y2[3..7]
    __shared__ float    sRBF[EWAVES][16][NRBF];
    __shared__ _Float16 sH[EWAVES][16][HID];
    __shared__ float    sWc[EWAVES][16][WCOLS];
    __shared__ int      sSrc[EWAVES][16];
    __shared__ int      sDst[EWAVES][16];

    const int wv = threadIdx.x >> 5;
    const int lane = threadIdx.x & 31;
    const int m = lane & 15;
    const int hf = lane >> 4;
    const int tileBase = (blockIdx.x * EWAVES + wv) * 16;

    // ---- per-edge geometry / spherical harmonics / RBF (lanes 0..15) ----
    if (lane < 16) {
        int e = tileBase + lane;
        float Y[8]; float rb[NRBF];
        int s = 0, dd = 0;
        if (e < E) {
            s  = eidx[e];
            dd = eidx[E + e];
            float rx = pos[3 * s + 0] - pos[3 * dd + 0];
            float ry = pos[3 * s + 1] - pos[3 * dd + 1];
            float rz = pos[3 * s + 2] - pos[3 * dd + 2];
            float d = sqrtf(rx * rx + ry * ry + rz * rz + 1e-12f);
            float inv = 1.0f / d;
            float x = rx * inv, y = ry * inv, z = rz * inv;
            Y[0] = x; Y[1] = y; Y[2] = z;
            Y[3] = x * y; Y[4] = y * z;
            Y[5] = (3.0f * z * z - 1.0f) * 0.28867513459481287f;  // /(2*sqrt(3))
            Y[6] = x * z; Y[7] = 0.5f * (x * x - y * y);
            #pragma unroll
            for (int k = 0; k < NRBF; ++k) {
                float c = (5.0f / 7.0f) * (float)k;  // linspace(0,5,8)
                float t = d - c;
                rb[k] = __expf(-t * t);
            }
        } else {
            #pragma unroll
            for (int k = 0; k < 8; ++k) { Y[k] = 0.0f; }
            #pragma unroll
            for (int k = 0; k < NRBF; ++k) { rb[k] = 0.0f; }
        }
        sSrc[wv][lane] = s; sDst[wv][lane] = dd;
        #pragma unroll
        for (int k = 0; k < 8; ++k) sY[wv][lane][k] = Y[k];
        #pragma unroll
        for (int k = 0; k < NRBF; ++k) sRBF[wv][lane][k] = rb[k];
    }

    // ---- A fragment for GEMM1 (16x32 f16, only K<8 nonzero) ----
    // lane layout: row M = m; elem i -> K = hf*16+i
    V16H a1;
    #pragma unroll
    for (int i = 0; i < 16; ++i) a1.e[i] = (_Float16)0.0f;
    if (hf == 0) {
        #pragma unroll
        for (int i = 0; i < 8; ++i) a1.e[i] = (_Float16)sRBF[wv][m][i];
    }

    // ---- GEMM1: h = silu(rbf @ Wr1 + br1), 4 N-tiles of 16 ----
    const v16h* B1 = (const v16h*)p1;
    #pragma unroll
    for (int t = 0; t < 4; ++t) {
        v8f acc = {0.f, 0.f, 0.f, 0.f, 0.f, 0.f, 0.f, 0.f};
        v16h b = B1[t * 32 + lane];
        acc = __builtin_amdgcn_wmma_f32_16x16x32_f16(false, a1.v, false, b,
                                                     (short)0, acc, false, false);
        int col = t * 16 + m;
        float bias = br1[col];
        #pragma unroll
        for (int r = 0; r < 8; ++r) {
            // C layout: lane holds col=m, rows hf*8+r
            sH[wv][hf * 8 + r][col] = (_Float16)siluf_(acc[r] + bias);
        }
    }

    // ---- rebuild A fragments for GEMM2 from h (K = 64 -> 2 chunks of 32) ----
    V16H a2_0, a2_1;
    #pragma unroll
    for (int i = 0; i < 8; ++i) {
        a2_0.e[i]     = sH[wv][m][hf * 16 + i];
        a2_0.e[8 + i] = sH[wv][m][hf * 16 + 8 + i];
        a2_1.e[i]     = sH[wv][m][32 + hf * 16 + i];
        a2_1.e[8 + i] = sH[wv][m][32 + hf * 16 + 8 + i];
    }

    // ---- GEMM2: w = h @ Wr2 (64x224), 14 N-tiles x 2 K-chunks ----
    const v16h* B2 = (const v16h*)p2;
    #pragma unroll
    for (int t = 0; t < 14; ++t) {
        v8f c = {0.f, 0.f, 0.f, 0.f, 0.f, 0.f, 0.f, 0.f};
        v16h b0 = B2[t * 32 + lane];
        c = __builtin_amdgcn_wmma_f32_16x16x32_f16(false, a2_0.v, false, b0,
                                                   (short)0, c, false, false);
        v16h b1 = B2[(14 + t) * 32 + lane];
        c = __builtin_amdgcn_wmma_f32_16x16x32_f16(false, a2_1.v, false, b1,
                                                   (short)0, c, false, false);
        int col = t * 16 + m;
        #pragma unroll
        for (int r = 0; r < 8; ++r) sWc[wv][hf * 8 + r][col] = c[r];
    }

    // ---- messages + atomic scatter (lane == channel l, loop over 16 edges) ----
    const float norm = 0.25f;  // 1/sqrt(AVG_DEG)
    for (int el = 0; el < 16; ++el) {
        int e = tileBase + el;       // uniform across the wave
        if (e >= E) break;
        int s  = sSrc[wv][el];
        int dd = sDst[wv][el];
        const float* xs = si1 + (size_t)s * DIM;
        float* od = conv + (size_t)dd * DIM;
        int l = lane;
        float x0 = xs[l];
        float x1v[3], x2v[5], Y1v[3], Y2v[5];
        #pragma unroll
        for (int c = 0; c < 3; ++c) { x1v[c] = xs[32 + l * 3 + c];  Y1v[c] = sY[wv][el][c]; }
        #pragma unroll
        for (int c = 0; c < 5; ++c) { x2v[c] = xs[128 + l * 5 + c]; Y2v[c] = sY[wv][el][3 + c]; }
        float w0 = sWc[wv][el][0 * 32 + l];
        float w1 = sWc[wv][el][1 * 32 + l];
        float w2 = sWc[wv][el][2 * 32 + l];
        float w3 = sWc[wv][el][3 * 32 + l];
        float w4 = sWc[wv][el][4 * 32 + l];
        float w5 = sWc[wv][el][5 * 32 + l];
        float w6 = sWc[wv][el][6 * 32 + l];
        float dot1 = x1v[0] * Y1v[0] + x1v[1] * Y1v[1] + x1v[2] * Y1v[2];
        float dot2 = x2v[0] * Y2v[0] + x2v[1] * Y2v[1] + x2v[2] * Y2v[2]
                   + x2v[3] * Y2v[3] + x2v[4] * Y2v[4];
        atomicAdd(od + l, (w0 * x0 + w4 * dot1 + w6 * dot2) * norm);
        #pragma unroll
        for (int c = 0; c < 3; ++c)
            atomicAdd(od + 32 + l * 3 + c, (w1 * x1v[c] + w3 * x0 * Y1v[c]) * norm);
        #pragma unroll
        for (int c = 0; c < 5; ++c)
            atomicAdd(od + 128 + l * 5 + c, (w2 * x2v[c] + w5 * x0 * Y2v[c]) * norm);
    }
}

// ---------------------------------------------------------------------------
extern "C" void kernel_launch(void* const* d_in, const int* in_sizes, int n_in,
                              void* d_out, int out_size, void* d_ws, size_t ws_size,
                              hipStream_t stream) {
    const float* nodes = (const float*)d_in[0];
    const float* pos   = (const float*)d_in[1];
    const float* W0    = (const float*)d_in[2];
    const float* W1    = (const float*)d_in[3];
    const float* W2    = (const float*)d_in[4];
    const float* Wr1   = (const float*)d_in[5];
    const float* br1   = (const float*)d_in[6];
    const float* Wr2   = (const float*)d_in[7];
    const float* Wg    = (const float*)d_in[8];
    // d_in[9] = batch (unused)
    const int*   eidx  = (const int*)d_in[10];

    int N = in_sizes[0] / DIM;
    int E = in_sizes[10] / 2;
    float* out = (float*)d_out;

    float* si1  = (float*)d_ws;
    float* conv = si1 + (size_t)N * DIM;
    _Float16* p1 = (_Float16*)(conv + (size_t)N * DIM);   // 4*32*16 halves
    _Float16* p2 = p1 + 4 * 32 * 16;                      // 28*32*16 halves

    // 1) pack MLP weights into f16 WMMA B-fragments
    pack_kernel<<<32, 32, 0, stream>>>(Wr1, Wr2, p1, p2);
    // 2) zero the scatter accumulator
    long long convN = (long long)N * DIM;
    zero_kernel<<<1024, 256, 0, stream>>>(conv, convN);
    // 3) si1 = irreps_linear(nodes)
    int nodeBlocks = (N + 7) / 8;
    linear_si_kernel<<<nodeBlocks, 256, 0, stream>>>(nodes, W0, W1, W2, si1, N);
    // 4) edge convolution with WMMA MLP + atomic scatter
    int edgeBlocks = (E + EWAVES * 16 - 1) / (EWAVES * 16);
    edge_kernel<<<edgeBlocks, EBLOCK, 0, stream>>>(si1, pos, eidx, br1, p1, p2, conv, E);
    // 5) out = gate(nodes + irreps_linear(conv))
    linear_gate_kernel<<<nodeBlocks, 256, 0, stream>>>(conv, nodes, W0, W1, W2, Wg, out, N);
}
